// MultiHeadAttention_44753559224412
// MI455X (gfx1250) — compile-verified, matrix operand order checked by reading, not run on hardware
//
#include <hip/hip_runtime.h>

// ---------------------------------------------------------------------------
// MI455X (gfx1250) causal multi-head attention forward.
//   B=4, N=2048, E=1024, H=16, D=64.  f32 in/out, f16 WMMA internally.
// Pipeline:  qkv_proj (f32->f16, V stored transposed)  ->  flash_attn
//            (TDM-staged K/V tiles, online softmax)     ->  out_proj (+bias).
// Workspace: Q(16MB) K(16MB) Vt(16MB) AO(16MB) = 64MB of d_ws.
// ---------------------------------------------------------------------------

typedef __attribute__((ext_vector_type(16))) _Float16 v16h;
typedef __attribute__((ext_vector_type(8)))  float    v8f;
typedef __attribute__((ext_vector_type(4)))  float    v4f;
typedef unsigned int u32x4 __attribute__((ext_vector_type(4)));
typedef int          i32x8 __attribute__((ext_vector_type(8)));
typedef int          i32x4 __attribute__((ext_vector_type(4)));

#define NB 4
#define NN 2048
#define NE 1024
#define NH 16
#define ND 64

__constant__ const float kScale = 0.125f;              // 1/sqrt(64)
__constant__ const float kLog2e = 1.4426950408889634f;

// D = A(16x32) x B(32x16) + C.  (neg_a, A, neg_b, B, c_mod, C, reuse_a, reuse_b)
__device__ __forceinline__ v8f wmma_f16(v16h a, v16h b, v8f c) {
  return __builtin_amdgcn_wmma_f32_16x16x32_f16(false, a, false, b,
                                                (short)0, c, false, false);
}

// A/B fragment from row-major f32 source: lane L -> row+(L%16), col+16*(L/16).
__device__ __forceinline__ v16h frag_from_f32(const float* __restrict__ base,
                                              int row, int col, int ld) {
  const int lane = threadIdx.x & 31;
  const float* p = base + (size_t)(row + (lane & 15)) * ld + col + ((lane >> 4) << 4);
  v4f f0 = *(const v4f*)(p + 0);
  v4f f1 = *(const v4f*)(p + 4);
  v4f f2 = *(const v4f*)(p + 8);
  v4f f3 = *(const v4f*)(p + 12);
  v16h h;
#pragma unroll
  for (int i = 0; i < 4; ++i) {
    h[i]      = (_Float16)f0[i];
    h[4 + i]  = (_Float16)f1[i];
    h[8 + i]  = (_Float16)f2[i];
    h[12 + i] = (_Float16)f3[i];
  }
  return h;
}

// Same fragment pattern from a row-major f16 source (global or LDS).
__device__ __forceinline__ v16h frag_from_f16(const _Float16* __restrict__ base,
                                              int row, int col, int ld) {
  const int lane = threadIdx.x & 31;
  return *(const v16h*)(base + (size_t)(row + (lane & 15)) * ld + col +
                        ((lane >> 4) << 4));
}

// Reductions across the 16 lanes that hold one C-layout row.
__device__ __forceinline__ float rowmax16(float v) {
  v = fmaxf(v, __shfl_xor(v, 1, 32));
  v = fmaxf(v, __shfl_xor(v, 2, 32));
  v = fmaxf(v, __shfl_xor(v, 4, 32));
  v = fmaxf(v, __shfl_xor(v, 8, 32));
  return v;
}
__device__ __forceinline__ float rowsum16(float v) {
  v += __shfl_xor(v, 1, 32);
  v += __shfl_xor(v, 2, 32);
  v += __shfl_xor(v, 4, 32);
  v += __shfl_xor(v, 8, 32);
  return v;
}

// ---------------------------------------------------------------------------
// Tensor Data Mover: async 2D tile (f16 elements) global -> LDS.
// D# built per CDNA5 ISA (group0 + group1, 2D tile;
// groups 2/3 and trailing group zero).  Tracked by TENSORcnt.
// NOTE: this toolchain exposes the 6-arg builtin
//   (uint32x4 g0, int32x8 g1, int32x4 g2, int32x4 g3, int32x8, i32 cpol).
// ---------------------------------------------------------------------------
__device__ __forceinline__ void tdm_load_2d_f16(
    unsigned int lds_byte_off, const _Float16* gptr, unsigned int tile_d0,
    unsigned int tile_d1, unsigned int tensor_d0, unsigned int tensor_d1,
    unsigned int stride_d0_elems) {
  const unsigned long long va = (unsigned long long)(const void*)gptr;
  u32x4 g0 = {};
  g0[0] = 1u;                                     // count=1, user descriptor
  g0[1] = lds_byte_off;                           // D#.lds_addr
  g0[2] = (unsigned int)va;                       // D#.global_addr[31:0]
  g0[3] = (unsigned int)(va >> 32) | (2u << 30);  // addr[56:32] | type=2
  i32x8 g1 = {};
  g1[0] = (int)(1u << 16);                        // data_size=1 (2 bytes)
  g1[1] = (int)((tensor_d0 & 0xffffu) << 16);     // tensor_dim0[15:0] @63:48
  g1[2] = (int)((tensor_d0 >> 16) | ((tensor_d1 & 0xffffu) << 16));
  g1[3] = (int)((tensor_d1 >> 16) | (tile_d0 << 16));  // tile_dim0 @127:112
  g1[4] = (int)tile_d1;                           // tile_dim1; tile_dim2=0
  g1[5] = (int)stride_d0_elems;                   // tensor_dim0_stride[31:0]
  g1[6] = 0;
  g1[7] = 0;
  i32x4 gz4 = {0, 0, 0, 0};
  i32x8 gz8 = {};
  __builtin_amdgcn_tensor_load_to_lds(g0, g1, gz4, gz4, gz8, 0);
}

// ---------------------------------------------------------------------------
// Kernel 1: Q/K/V projection.  Y[m][f] = sum_k X[m][k] * W[f][k]  (x @ W.T)
//   blockIdx.z: 0->Q, 1->K (store [B,H,N,D] f16), 2->V (store [B,H,D,N] f16).
//   256 thr = 8 waves (4x2); wave tile 32x32; block tile 128(M) x 64(F).
// ---------------------------------------------------------------------------
__global__ __launch_bounds__(256) void qkv_proj_kernel(
    const float* __restrict__ xq, const float* __restrict__ xk,
    const float* __restrict__ xv, const float* __restrict__ Wq,
    const float* __restrict__ Wk, const float* __restrict__ Wv,
    _Float16* __restrict__ qout, _Float16* __restrict__ kout,
    _Float16* __restrict__ vtout) {
  const int which = blockIdx.z;
  const float* X = (which == 0) ? xq : (which == 1) ? xk : xv;
  const float* W = (which == 0) ? Wq : (which == 1) ? Wk : Wv;

  const int wave = threadIdx.x >> 5;
  const int lane = threadIdx.x & 31;
  const int wm = wave & 3, wf = wave >> 2;
  const int m0 = blockIdx.x * 128 + wm * 32;
  const int f0 = blockIdx.y * 64 + wf * 32;

  v8f c00 = {}, c01 = {}, c10 = {}, c11 = {};
#pragma unroll 4
  for (int k = 0; k < NE; k += 32) {
    if (k + 32 < NE) {  // stream-ahead hint: global_prefetch_b8
      __builtin_prefetch((const void*)(X + (size_t)(m0 + (lane & 15)) * NE + k + 32), 0, 3);
      __builtin_prefetch((const void*)(W + (size_t)(f0 + (lane & 15)) * NE + k + 32), 0, 3);
    }
    v16h a0 = frag_from_f32(X, m0, k, NE);
    v16h a1 = frag_from_f32(X, m0 + 16, k, NE);
    v16h b0 = frag_from_f32(W, f0, k, NE);
    v16h b1 = frag_from_f32(W, f0 + 16, k, NE);
    c00 = wmma_f16(a0, b0, c00);
    c01 = wmma_f16(a0, b1, c01);
    c10 = wmma_f16(a1, b0, c10);
    c11 = wmma_f16(a1, b1, c11);
  }

  v8f acc[2][2] = {{c00, c01}, {c10, c11}};
#pragma unroll
  for (int i = 0; i < 2; ++i)
#pragma unroll
    for (int j = 0; j < 2; ++j)
#pragma unroll
      for (int r = 0; r < 8; ++r) {
        const int m = m0 + 16 * i + r + 8 * (lane >> 4);  // global token row
        const int f = f0 + 16 * j + (lane & 15);          // feature column
        const int b = m / NN, n = m % NN;
        const int h = f / ND, d = f % ND;
        const _Float16 val = (_Float16)acc[i][j][r];
        if (which == 0)
          qout[(((size_t)(b * NH + h)) * NN + n) * ND + d] = val;
        else if (which == 1)
          kout[(((size_t)(b * NH + h)) * NN + n) * ND + d] = val;
        else
          vtout[(((size_t)(b * NH + h)) * ND + d) * NN + n] = val;  // V^T
      }
}

// ---------------------------------------------------------------------------
// Kernel 2: flash attention.  128 thr = 4 waves; each wave owns a 16-row
// q-tile; block covers 64 q rows of one (b,h).  K (32x64) and V^T (64x32)
// tiles are staged into block-shared LDS by the Tensor Data Mover, double
// buffered; the block runs a uniform k-tile count so __syncthreads() is the
// publish/retire barrier, and each wave predicates compute on its causal
// range.
// ---------------------------------------------------------------------------
__global__ __launch_bounds__(128) void flash_attn_kernel(
    const _Float16* __restrict__ Q, const _Float16* __restrict__ K,
    const _Float16* __restrict__ Vt, _Float16* __restrict__ AO) {
  // Halves: kbuf[2][32*64] | vbuf[2][64*32] | pbuf[4][16*32]  (20 KB total)
  __shared__ _Float16 smem[2 * 2048 + 2 * 2048 + 4 * 512];
  constexpr unsigned kOffH = 0;
  constexpr unsigned vOffH = 2 * 2048;
  constexpr unsigned pOffH = 4 * 2048;

  const int wave = threadIdx.x >> 5;
  const int lane = threadIdx.x & 31;
  const int b = blockIdx.z, h = blockIdx.y;
  const int qBase = blockIdx.x * 64 + wave * 16;
  const int qLast = qBase + 15;

  const _Float16* Qh = Q + ((size_t)(b * NH + h)) * NN * ND;
  const _Float16* Kh = K + ((size_t)(b * NH + h)) * NN * ND;
  const _Float16* Vh = Vt + ((size_t)(b * NH + h)) * ND * NN;

  // Q fragments: 16 rows x D=64 -> two K-steps of 32.
  v16h qf0 = frag_from_f16(Qh, qBase, 0, ND);
  v16h qf1 = frag_from_f16(Qh, qBase, 32, ND);

  float mrow[8], lrow[8];
  v8f o0 = {}, o1 = {}, o2 = {}, o3 = {};
#pragma unroll
  for (int r = 0; r < 8; ++r) { mrow[r] = -1e30f; lrow[r] = 0.0f; }

  // Uniform per-block trip count (covers qBase = blockIdx.x*64 + 48..63).
  const int nkt = blockIdx.x * 2 + 2;

  if (threadIdx.x < 32) {  // wave 0 drives the TDM
    tdm_load_2d_f16(kOffH * 2, Kh, /*tile*/ 64, 32, /*tensor*/ ND, NN, ND);
    tdm_load_2d_f16(vOffH * 2, Vh, /*tile*/ 32, 64, /*tensor*/ NN, ND, NN);
  }

  for (int kt = 0; kt < nkt; ++kt) {
    const int n0 = kt * 32;
    const int cur = kt & 1;
    if (threadIdx.x < 32) {
      if (kt + 1 < nkt) {
        const unsigned nxt = (unsigned)(cur ^ 1);
        tdm_load_2d_f16((kOffH + nxt * 2048) * 2, Kh + (size_t)(n0 + 32) * ND,
                        64, 32, ND, NN, ND);
        tdm_load_2d_f16((vOffH + nxt * 2048) * 2, Vh + (n0 + 32),
                        32, 64, NN, ND, NN);
        __builtin_amdgcn_s_wait_tensorcnt(2);  // current pair complete
      } else {
        __builtin_amdgcn_s_wait_tensorcnt(0);
      }
    }
    __syncthreads();  // publish buffer `cur`

    if (n0 <= qLast) {  // this wave's causal range
      const _Float16* kb = &smem[kOffH + (unsigned)cur * 2048];  // [32][64]
      const _Float16* vb = &smem[vOffH + (unsigned)cur * 2048];  // [64][32]
      _Float16* pb = &smem[pOffH + (unsigned)wave * 512];        // [16][32]

      // S = scale * Q K^T : 16 x 32 -> two 16x16 accumulators.
      v8f s0 = {}, s1 = {};
      s0 = wmma_f16(qf0, frag_from_f16(kb, 0, 0, ND), s0);
      s0 = wmma_f16(qf1, frag_from_f16(kb, 0, 32, ND), s0);
      s1 = wmma_f16(qf0, frag_from_f16(kb, 16, 0, ND), s1);
      s1 = wmma_f16(qf1, frag_from_f16(kb, 16, 32, ND), s1);

      // Scale + causal mask + online softmax.
      const int col0 = n0 + (lane & 15);
      const int col1 = n0 + 16 + (lane & 15);
#pragma unroll
      for (int r = 0; r < 8; ++r) {
        const int row = qBase + r + 8 * (lane >> 4);
        float a = s0[r] * kScale;
        float c = s1[r] * kScale;
        if (col0 > row) a = -1e30f;
        if (col1 > row) c = -1e30f;
        const float mnew = fmaxf(mrow[r], rowmax16(fmaxf(a, c)));
        const float alpha = exp2f((mrow[r] - mnew) * kLog2e);
        const float p0 = exp2f((a - mnew) * kLog2e);
        const float p1 = exp2f((c - mnew) * kLog2e);
        o0[r] *= alpha; o1[r] *= alpha; o2[r] *= alpha; o3[r] *= alpha;
        lrow[r] = lrow[r] * alpha + rowsum16(p0 + p1);
        mrow[r] = mnew;
        // Stash P in LDS: C-layout -> A-layout for the PV WMMA.
        pb[(r + 8 * (lane >> 4)) * 32 + (lane & 15)] = (_Float16)p0;
        pb[(r + 8 * (lane >> 4)) * 32 + 16 + (lane & 15)] = (_Float16)p1;
      }
      asm volatile("s_wait_dscnt 0" ::: "memory");
      const v16h pA = *(const v16h*)(&pb[(lane & 15) * 32 + ((lane >> 4) << 4)]);

      // O += P V : B operand is contiguous rows of the staged V^T tile.
      o0 = wmma_f16(pA, frag_from_f16(vb, 0, 0, 32), o0);
      o1 = wmma_f16(pA, frag_from_f16(vb, 16, 0, 32), o1);
      o2 = wmma_f16(pA, frag_from_f16(vb, 32, 0, 32), o2);
      o3 = wmma_f16(pA, frag_from_f16(vb, 48, 0, 32), o3);
    }
    __syncthreads();  // all readers done with `cur` before it is re-staged
  }

  // Normalize and store merged-head AO[b][n][h*D + d] (f16).
#pragma unroll
  for (int r = 0; r < 8; ++r) {
    const float inv = 1.0f / lrow[r];
    const int row = qBase + r + 8 * (lane >> 4);
    const size_t base = ((size_t)b * NN + row) * NE + h * ND + (lane & 15);
    AO[base + 0]  = (_Float16)(o0[r] * inv);
    AO[base + 16] = (_Float16)(o1[r] * inv);
    AO[base + 32] = (_Float16)(o2[r] * inv);
    AO[base + 48] = (_Float16)(o3[r] * inv);
  }
}

// ---------------------------------------------------------------------------
// Kernel 3: output projection  out = AO @ Wo.T + bo   (f32 out).
// ---------------------------------------------------------------------------
__global__ __launch_bounds__(256) void out_proj_kernel(
    const _Float16* __restrict__ AO, const float* __restrict__ Wo,
    const float* __restrict__ bo, float* __restrict__ out) {
  const int wave = threadIdx.x >> 5;
  const int lane = threadIdx.x & 31;
  const int wm = wave & 3, wf = wave >> 2;
  const int m0 = blockIdx.x * 128 + wm * 32;
  const int f0 = blockIdx.y * 64 + wf * 32;

  v8f c00 = {}, c01 = {}, c10 = {}, c11 = {};
#pragma unroll 4
  for (int k = 0; k < NE; k += 32) {
    if (k + 32 < NE) {
      __builtin_prefetch((const void*)(AO + (size_t)(m0 + (lane & 15)) * NE + k + 32), 0, 3);
      __builtin_prefetch((const void*)(Wo + (size_t)(f0 + (lane & 15)) * NE + k + 32), 0, 3);
    }
    v16h a0 = frag_from_f16(AO, m0, k, NE);
    v16h a1 = frag_from_f16(AO, m0 + 16, k, NE);
    v16h b0 = frag_from_f32(Wo, f0, k, NE);
    v16h b1 = frag_from_f32(Wo, f0 + 16, k, NE);
    c00 = wmma_f16(a0, b0, c00);
    c01 = wmma_f16(a0, b1, c01);
    c10 = wmma_f16(a1, b0, c10);
    c11 = wmma_f16(a1, b1, c11);
  }

  v8f acc[2][2] = {{c00, c01}, {c10, c11}};
#pragma unroll
  for (int i = 0; i < 2; ++i)
#pragma unroll
    for (int j = 0; j < 2; ++j) {
      const int f = f0 + 16 * j + (lane & 15);
      const float bias = bo[f];
#pragma unroll
      for (int r = 0; r < 8; ++r) {
        const int m = m0 + 16 * i + r + 8 * (lane >> 4);
        out[(size_t)m * NE + f] = acc[i][j][r] + bias;
      }
    }
}

// ---------------------------------------------------------------------------
extern "C" void kernel_launch(void* const* d_in, const int* in_sizes, int n_in,
                              void* d_out, int out_size, void* d_ws,
                              size_t ws_size, hipStream_t stream) {
  const float* xq = (const float*)d_in[0];
  const float* xk = (const float*)d_in[1];
  const float* xv = (const float*)d_in[2];
  // d_in[3] = attn_mask (bool) — causal, recomputed analytically in-kernel.
  const float* Wq = (const float*)d_in[4];
  const float* Wk = (const float*)d_in[5];
  const float* Wv = (const float*)d_in[6];
  const float* Wo = (const float*)d_in[7];
  const float* bo = (const float*)d_in[8];
  float* out = (float*)d_out;

  const size_t elems = (size_t)NB * NN * NE;  // 8M f16 per tensor
  _Float16* qws = (_Float16*)d_ws;
  _Float16* kws = qws + elems;
  _Float16* vtws = kws + elems;
  _Float16* aows = vtws + elems;

  dim3 pgrid(NB * NN / 128, NE / 64, 3);
  qkv_proj_kernel<<<pgrid, dim3(256), 0, stream>>>(xq, xk, xv, Wq, Wk, Wv, qws,
                                                   kws, vtws);

  dim3 agrid(NN / 64, NH, NB);
  flash_attn_kernel<<<agrid, dim3(128), 0, stream>>>(qws, kws, vtws, aows);

  dim3 ogrid(NB * NN / 128, NE / 64);
  out_proj_kernel<<<ogrid, dim3(256), 0, stream>>>(aows, Wo, bo, out);
}
